// node_update_layer_51427938402670
// MI455X (gfx1250) — matrix operand order repaired and checked by reading, not compiled, not run on hardware
//
#include <hip/hip_runtime.h>
#include <math.h>

// ---------------------------------------------------------------------------
// MI455X (gfx1250) implementation.
// Heavy math (2 batched einsum-GEMMs + 4 fused 2-layer MLPs) runs on
// V_WMMA_F32_16X16X32_BF16 (fp32 accumulate, bf16 operands converted at LDS
// staging). wave32; 256-thread blocks = 8 waves; each wave owns 16x16 tiles.
// Fragments load as two aligned ds_load_b128 per operand; all transposed
// staging converts float PAIRS (one v_cvt_pk_bf16_f32) and stores b32, except the eu W2 head (scalar bf16 stores).
// ---------------------------------------------------------------------------

typedef __attribute__((ext_vector_type(16))) __bf16 v16bf;
typedef __attribute__((ext_vector_type(4)))  __bf16 v4bf;
typedef __attribute__((ext_vector_type(2)))  __bf16 v2bf;
typedef __attribute__((ext_vector_type(8)))  float  v8f;
typedef __attribute__((ext_vector_type(4)))  float  v4f;
typedef __attribute__((ext_vector_type(2)))  float  v2f;
typedef __attribute__((ext_vector_type(4)))  unsigned int v4u;

#define DEV_INLINE static __device__ __forceinline__

constexpr int B_ = 256, K_ = 512, L_ = 512, D_ = 64, CH_ = 128;
constexpr int OUTW = CH_ + D_;   // 192

DEV_INLINE v8f wmma_bf16(v16bf a, v16bf b, v8f c) {
  return __builtin_amdgcn_wmma_f32_16x16x32_bf16(
      false, a, false, b, (short)0, c, false, false);
}

// A fragment (16x32 bf16, row-major LDS tile, stride multiple of 8 elems).
DEV_INLINE v16bf load_a_frag(const __bf16* base, int stride) {
  int lane = threadIdx.x & 31;
  int m = lane & 15, hl = lane >> 4;
  const __bf16* p = base + m * stride + hl * 8;
  union { v4u u[2]; v16bf f; } u;
  u.u[0] = *(const v4u*)p;
  u.u[1] = *(const v4u*)(p + 16);
  return u.f;
}

// B fragment from TRANSPOSED tile Bt[n][k] (strideT multiple of 8 elems).
DEV_INLINE v16bf load_bT_frag(const __bf16* baseT, int strideT) {
  int lane = threadIdx.x & 31;
  int n = lane & 15, hl = lane >> 4;
  const __bf16* p = baseT + n * strideT + hl * 8;
  union { v4u u[2]; v16bf f; } u;
  u.u[0] = *(const v4u*)p;
  u.u[1] = *(const v4u*)(p + 16);
  return u.f;
}

DEV_INLINE v8f vzero() { v8f z = {0.f,0.f,0.f,0.f,0.f,0.f,0.f,0.f}; return z; }
DEV_INLINE float frcp(float x) { return __builtin_amdgcn_rcpf(x); }
DEV_INLINE v2bf pack2(float a, float b) { v2bf p; p[0] = (__bf16)a; p[1] = (__bf16)b; return p; }

// ---------------------------------------------------------------------------
// L1 sums of |e| along L (colsum, per (b,k)) and along K (rowsum, per (b,l)).
// ---------------------------------------------------------------------------
__global__ __launch_bounds__(256) void rowsum_kernel(const float* __restrict__ e,
                                                     float* __restrict__ rowsum) {
  int w = threadIdx.x >> 5, lane = threadIdx.x & 31;
  int row = blockIdx.x * 8 + w;                 // over B*L
  const float* p = e + (size_t)row * K_;
  float s = 0.f;
  for (int k = lane; k < K_; k += 32) s += fabsf(p[k]);
#pragma unroll
  for (int off = 16; off; off >>= 1) s += __shfl_xor(s, off, 32);
  if (lane == 0) rowsum[row] = s;
}

__global__ __launch_bounds__(256) void colsum_kernel(const float* __restrict__ e,
                                                     float* __restrict__ colsum) {
  int gid = blockIdx.x * 256 + threadIdx.x;     // over B*K
  int b = gid >> 9, k = gid & (K_ - 1);
  const float* p = e + (size_t)b * L_ * K_ + k;
  float s = 0.f;
  for (int l = 0; l < L_; ++l) s += fabsf(p[(size_t)l * K_]);
  colsum[gid] = s;
}

__global__ __launch_bounds__(256) void top2_kernel(const float* __restrict__ uk,
                                                   float* __restrict__ max1,
                                                   float* __restrict__ max2) {
  int gid = blockIdx.x * 256 + threadIdx.x;     // over B*D
  int b = gid >> 6, d = gid & 63;
  const float* p = uk + (size_t)b * K_ * D_ + d;
  float m1 = -3.4e38f, m2 = -3.4e38f;
  for (int k = 0; k < K_; ++k) {
    float v = p[(size_t)k * D_];
    if (v > m1) { m2 = m1; m1 = v; } else if (v > m2) { m2 = v; }
  }
  max1[gid] = m1; max2[gid] = m2;
}

__global__ __launch_bounds__(256) void mean_uc_kernel(const float* __restrict__ uk,
                                                      const float* __restrict__ erc,
                                                      float* __restrict__ mean_uc) {
  int gid = blockIdx.x * 256 + threadIdx.x;     // over B*D
  int b = gid >> 6, d = gid & 63;
  const float* p = uk + (size_t)b * K_ * D_ + d;
  const float* w = erc + (size_t)b * K_;
  float s = 0.f;
  for (int k = 0; k < K_; ++k) s += p[(size_t)k * D_] * w[k];
  mean_uc[gid] = s;
}

// ---------------------------------------------------------------------------
// Batched mean GEMM: C[b, r, 0:64] = rcp(max(sums[b,r],1e-12)) * sum_l A*B
//   TRANSA=1: A[r,l] = e[b,l,r]   (mean_uk: e^T @ rl, scaled by colsum)
//   TRANSA=0: A[r,l] = e[b,r,l]   (mean_rl: e   @ uk, scaled by rowsum)
// ---------------------------------------------------------------------------
template <bool TRANSA>
__global__ __launch_bounds__(256) void mean_gemm(const float* __restrict__ E,
                                                 const float* __restrict__ Bm,
                                                 const float* __restrict__ sums,
                                                 float* __restrict__ C) {
  constexpr int SA = 40, SBT = 40;
  __shared__ __align__(16) __bf16 As[64 * SA];        // [r][l] 64x32
  __shared__ __align__(16) __bf16 Bt[64 * SBT];       // [d][l] 64x32 (transposed)
  int t = threadIdx.x, w = t >> 5, lane = t & 31;
  int b = blockIdx.y, r0 = blockIdx.x * 64;
  const float* Eb = E + (size_t)b * L_ * K_;
  const float* Bb = Bm + (size_t)b * L_ * D_;
  const float* sb = sums + (size_t)b * K_;            // K_ == L_ == 512
  float* Cb = C + (size_t)b * K_ * D_;
  int tr = w & 3, tc0 = (w >> 2) * 2;
  int x4 = t & 15, cp = t >> 4;                       // 16 quads x 16 col-pairs
  v8f acc0 = vzero(), acc1 = vzero();

  for (int l0 = 0; l0 < L_; l0 += 32) {
    __syncthreads();
    if (TRANSA) {
      // A[r][c] = Eb[(l0+c)*K + r0+r]; pair adjacent c, pack bf16 pairs.
      v4f fa = *(const v4f*)&Eb[(size_t)(l0 + 2 * cp) * K_ + (r0 + x4 * 4)];
      v4f fb = *(const v4f*)&Eb[(size_t)(l0 + 2 * cp + 1) * K_ + (r0 + x4 * 4)];
      v4f ga = *(const v4f*)&Bb[(size_t)(l0 + 2 * cp) * D_ + x4 * 4];
      v4f gb = *(const v4f*)&Bb[(size_t)(l0 + 2 * cp + 1) * D_ + x4 * 4];
#pragma unroll
      for (int q = 0; q < 4; ++q) {
        *(v2bf*)&As[(x4 * 4 + q) * SA + 2 * cp] = pack2(fa[q], fb[q]);
        *(v2bf*)&Bt[(x4 * 4 + q) * SBT + 2 * cp] = pack2(ga[q], gb[q]);
      }
    } else {
      // A[r][c] = Eb[(r0+r)*K + l0+c]; contiguous c -> v4bf stores.
      v4f f0 = *(const v4f*)&Eb[(size_t)(r0 + (t >> 3)) * K_ + (l0 + (t & 7) * 4)];
      v4f f1 = *(const v4f*)&Eb[(size_t)(r0 + 32 + (t >> 3)) * K_ + (l0 + (t & 7) * 4)];
      v4f ga = *(const v4f*)&Bb[(size_t)(l0 + 2 * cp) * D_ + x4 * 4];
      v4f gb = *(const v4f*)&Bb[(size_t)(l0 + 2 * cp + 1) * D_ + x4 * 4];
      v4bf h0; v4bf h1;
#pragma unroll
      for (int q = 0; q < 4; ++q) { h0[q] = (__bf16)f0[q]; h1[q] = (__bf16)f1[q]; }
      *(v4bf*)&As[(t >> 3) * SA + (t & 7) * 4] = h0;
      *(v4bf*)&As[(32 + (t >> 3)) * SA + (t & 7) * 4] = h1;
#pragma unroll
      for (int q = 0; q < 4; ++q)
        *(v2bf*)&Bt[(x4 * 4 + q) * SBT + 2 * cp] = pack2(ga[q], gb[q]);
    }
    __syncthreads();
    v16bf af  = load_a_frag(As + tr * 16 * SA, SA);
    v16bf bf0 = load_bT_frag(Bt + (tc0 * 16) * SBT, SBT);
    v16bf bf1 = load_bT_frag(Bt + ((tc0 + 1) * 16) * SBT, SBT);
    acc0 = wmma_bf16(af, bf0, acc0);
    acc1 = wmma_bf16(af, bf1, acc1);
  }

  int hl = lane >> 4, n = lane & 15;
#pragma unroll
  for (int r = 0; r < 8; ++r) {
    int m = r + 8 * hl;
    int row = r0 + tr * 16 + m;
    float s = frcp(fmaxf(sb[row], 1e-12f));
    Cb[(size_t)row * D_ + tc0 * 16 + n]       = acc0[r] * s;
    Cb[(size_t)row * D_ + (tc0 + 1) * 16 + n] = acc1[r] * s;
  }
}

// ---------------------------------------------------------------------------
// Fused 2-layer MLP: out = relu(X@W1 + b1)@W2 + b2, hidden kept in LDS (bf16).
// MODE 0: fu — [uk | loo_max | mean_uk]  (CIN=192)
// MODE 1: f  — [rl | mean_rl]            (CIN=128)
// MODE 2: fc — [mean_uc | uc]            (CIN=128)
// MODE 3: eu — |uc_up[b,:] - uk_up[b,k,:]| (CIN=192)
// ---------------------------------------------------------------------------
struct MlpP {
  const float* s0; const float* s1;
  const float* m1; const float* m2;
  const float* w1; const float* b1;
  const float* w2; const float* b2;
  float* out; int ostride; int coutv;
};

template <int MODE>
DEV_INLINE float mlp_fetch(const MlpP& P, int row, int c) {
  if (MODE == 0) {
    if (c < 64) return P.s0[(size_t)row * 64 + c];
    if (c < 128) {
      int d = c - 64, b = row >> 9;
      float u = P.s0[(size_t)row * 64 + d];
      float a = P.m1[(size_t)b * 64 + d];
      float s = P.m2[(size_t)b * 64 + d];
      return (u == a) ? s : a;                      // leave-one-out max
    }
    return P.s1[(size_t)row * 64 + (c - 128)];
  } else if (MODE == 1 || MODE == 2) {
    return (c < 64) ? P.s0[(size_t)row * 64 + c]
                    : P.s1[(size_t)row * 64 + (c - 64)];
  } else {
    int b = row >> 9;
    return fabsf(P.s0[(size_t)b * OUTW + c] - P.s1[(size_t)row * OUTW + c]);
  }
}

template <int MODE, int CIN, int HID, int COUTP>
__global__ __launch_bounds__(256) void mlp2(MlpP P) {
  constexpr int SA = 40, SW1T = 40, SW2T = 40, SH = HID + 8;  // SH mult of 8
  // Stage-1 region (A tile + W1t) aliases stage-2 region (W2t).
  constexpr int S1B = (64 * SA + HID * SW1T) * 2;
  constexpr int S2B = (COUTP * SW2T) * 2;
  constexpr int SXB = S1B > S2B ? S1B : S2B;
  __shared__ __align__(16) char smemx[SXB];
  __shared__ __align__(16) __bf16 Hs[64 * SH];
  __bf16* As  = (__bf16*)smemx;                 // [r][k] 64x32
  __bf16* W1t = (__bf16*)smemx + 64 * SA;       // [h][k] HIDx32 (transposed)
  __bf16* W2t = (__bf16*)smemx;                 // [c][k] COUTPx32 (transposed)

  int t = threadIdx.x, w = t >> 5, lane = t & 31;
  int row0 = blockIdx.x * 64;
  int hl = lane >> 4, n = lane & 15;
  int tr = w & 3, ch = w >> 2;

  // ---- stage 1: H = relu(X @ W1 + b1); wave covers cols [ch*HID/2, +HID/2)
  constexpr int NT1 = HID / 32;
  v8f acc1[NT1];
#pragma unroll
  for (int i = 0; i < NT1; ++i) acc1[i] = vzero();

  for (int k0 = 0; k0 < CIN; k0 += 32) {
    __syncthreads();
    {   // X tile 64x32 gathered: 4 col-pairs per thread; load then pack-store.
      v2f va[4];
#pragma unroll
      for (int i = 0; i < 4; ++i) {
        int idx = i * 256 + t;
        int c2 = idx & 15, r = idx >> 4;
        va[i][0] = mlp_fetch<MODE>(P, row0 + r, k0 + c2 * 2);
        va[i][1] = mlp_fetch<MODE>(P, row0 + r, k0 + c2 * 2 + 1);
      }
#pragma unroll
      for (int i = 0; i < 4; ++i) {
        int idx = i * 256 + t;
        int c2 = idx & 15, r = idx >> 4;
        *(v2bf*)&As[r * SA + c2 * 2] = pack2(va[i][0], va[i][1]);
      }
    }
    {   // W1t tile HIDx32: pair adjacent k rows, pack bf16 pairs.
      constexpr int W1P = HID * 4 / 256;        // pair-items per thread
      v4f la[W1P], lb[W1P];
#pragma unroll
      for (int i = 0; i < W1P; ++i) {
        int idx = i * 256 + t;
        int h4 = idx % (HID / 4), kp = idx / (HID / 4);   // kp in 0..15
        la[i] = *(const v4f*)&P.w1[(size_t)(k0 + 2 * kp) * HID + h4 * 4];
        lb[i] = *(const v4f*)&P.w1[(size_t)(k0 + 2 * kp + 1) * HID + h4 * 4];
      }
#pragma unroll
      for (int i = 0; i < W1P; ++i) {
        int idx = i * 256 + t;
        int h4 = idx % (HID / 4), kp = idx / (HID / 4);
#pragma unroll
        for (int q = 0; q < 4; ++q)
          *(v2bf*)&W1t[(h4 * 4 + q) * SW1T + 2 * kp] = pack2(la[i][q], lb[i][q]);
      }
    }
    __syncthreads();
    v16bf af = load_a_frag(As + tr * 16 * SA, SA);
#pragma unroll
    for (int j = 0; j < NT1; ++j) {
      v16bf bf = load_bT_frag(W1t + (ch * (HID / 2) + j * 16) * SW1T, SW1T);
      acc1[j] = wmma_bf16(af, bf, acc1[j]);
    }
  }
  __syncthreads();
#pragma unroll
  for (int j = 0; j < NT1; ++j) {
#pragma unroll
    for (int r = 0; r < 8; ++r) {
      int m = r + 8 * hl;
      int h = ch * (HID / 2) + j * 16 + n;
      float v = acc1[j][r] + P.b1[h];
      Hs[(tr * 16 + m) * SH + h] = (__bf16)fmaxf(v, 0.0f);
    }
  }
  __syncthreads();

  // ---- stage 2: out = H @ W2 + b2  (W2 zero-padded to COUTP columns)
  constexpr int NCT = COUTP / 16;
  constexpr int NT2 = (NCT >= 2) ? NCT / 2 : 1;
  bool active = (NCT >= 2) || (ch == 0);
  int start = (NCT >= 2) ? ch * NT2 : 0;
  v8f acc2[NT2];
#pragma unroll
  for (int i = 0; i < NT2; ++i) acc2[i] = vzero();

  for (int k0 = 0; k0 < HID; k0 += 32) {
    __syncthreads();
    if (COUTP >= 64) {                          // dense head: paired pack
      constexpr int W2P = (COUTP >= 64) ? (COUTP * 4 / 256) : 1;
      v4f la[W2P], lb[W2P];
#pragma unroll
      for (int i = 0; i < W2P; ++i) {
        int idx = i * 256 + t;
        int c4 = idx % (COUTP / 4), kp = idx / (COUTP / 4);
        la[i] = *(const v4f*)&P.w2[(size_t)(k0 + 2 * kp) * COUTP + c4 * 4];
        lb[i] = *(const v4f*)&P.w2[(size_t)(k0 + 2 * kp + 1) * COUTP + c4 * 4];
      }
#pragma unroll
      for (int i = 0; i < W2P; ++i) {
        int idx = i * 256 + t;
        int c4 = idx % (COUTP / 4), kp = idx / (COUTP / 4);
#pragma unroll
        for (int q = 0; q < 4; ++q)
          *(v2bf*)&W2t[(c4 * 4 + q) * SW2T + 2 * kp] = pack2(la[i][q], lb[i][q]);
      }
    } else {                                    // eu head: scalar, zero-pad
      for (int idx = t; idx < 32 * COUTP; idx += 256) {
        int c = idx % COUTP, kk = idx / COUTP;
        float v = (c < P.coutv) ? P.w2[(size_t)(k0 + kk) * P.coutv + c] : 0.0f;
        W2t[c * SW2T + kk] = (__bf16)v;
      }
    }
    __syncthreads();
    if (active) {
      v16bf af = load_a_frag(Hs + (tr * 16) * SH + k0, SH);
#pragma unroll
      for (int j = 0; j < NT2; ++j) {
        v16bf bf = load_bT_frag(W2t + ((start + j) * 16) * SW2T, SW2T);
        acc2[j] = wmma_bf16(af, bf, acc2[j]);
      }
    }
  }
  if (active) {
#pragma unroll
    for (int j = 0; j < NT2; ++j) {
#pragma unroll
      for (int r = 0; r < 8; ++r) {
        int m = r + 8 * hl;
        int c = (start + j) * 16 + n;
        int row = row0 + tr * 16 + m;
        if (c < P.coutv)
          P.out[(size_t)row * P.ostride + c] = acc2[j][r] + P.b2[c];
      }
    }
  }
}

// Copy a [rows,64] tensor into columns [col_off, col_off+64) of a strided dst.
__global__ __launch_bounds__(256) void concat_copy_kernel(const float* __restrict__ src,
                                                          float* __restrict__ dst,
                                                          int dst_stride, int col_off,
                                                          int total_half) {
  int gid = blockIdx.x * 256 + threadIdx.x;       // over total/2 (float2)
  if (gid >= total_half) return;
  int r = gid >> 5, c2 = gid & 31;
  const float* s = src + (size_t)r * 64 + c2 * 2;
  float* d = dst + (size_t)r * dst_stride + col_off + c2 * 2;
  d[0] = s[0]; d[1] = s[1];
}

// Softmax over K=512 per batch; 256 threads, 2 elements each.
__global__ __launch_bounds__(256) void softmax_kernel(const float* __restrict__ scores,
                                                      float* __restrict__ out) {
  __shared__ float red[256];
  int b = blockIdx.x, t = threadIdx.x;
  const float* s = scores + (size_t)b * K_;
  float v0 = s[t], v1 = s[t + 256];
  red[t] = fmaxf(v0, v1);
  __syncthreads();
  for (int off = 128; off; off >>= 1) {
    if (t < off) red[t] = fmaxf(red[t], red[t + off]);
    __syncthreads();
  }
  float m = red[0];
  __syncthreads();
  float e0 = __expf(v0 - m), e1 = __expf(v1 - m);
  red[t] = e0 + e1;
  __syncthreads();
  for (int off = 128; off; off >>= 1) {
    if (t < off) red[t] += red[t + off];
    __syncthreads();
  }
  float inv = frcp(red[0]);
  out[(size_t)b * K_ + t]       = e0 * inv;
  out[(size_t)b * K_ + t + 256] = e1 * inv;
}

// ---------------------------------------------------------------------------
extern "C" void kernel_launch(void* const* d_in, const int* in_sizes, int n_in,
                              void* d_out, int out_size, void* d_ws, size_t ws_size,
                              hipStream_t stream) {
  const float* uk    = (const float*)d_in[0];
  const float* uc    = (const float*)d_in[1];
  const float* rl    = (const float*)d_in[2];
  const float* e     = (const float*)d_in[3];
  const float* erc   = (const float*)d_in[4];
  const float* fu_w1 = (const float*)d_in[5];
  const float* fu_b1 = (const float*)d_in[6];
  const float* fu_w2 = (const float*)d_in[7];
  const float* fu_b2 = (const float*)d_in[8];
  const float* f_w1  = (const float*)d_in[9];
  const float* f_b1  = (const float*)d_in[10];
  const float* f_w2  = (const float*)d_in[11];
  const float* f_b2  = (const float*)d_in[12];
  const float* fc_w1 = (const float*)d_in[13];
  const float* fc_b1 = (const float*)d_in[14];
  const float* fc_w2 = (const float*)d_in[15];
  const float* fc_b2 = (const float*)d_in[16];
  const float* eu_w1 = (const float*)d_in[17];
  const float* eu_b1 = (const float*)d_in[18];
  const float* eu_w2 = (const float*)d_in[19];
  const float* eu_b2 = (const float*)d_in[20];

  float* out = (float*)d_out;
  float* uk_up  = out;                                        // [B,K,192]
  float* uc_up  = uk_up + (size_t)B_ * K_ * OUTW;             // [B,192]
  float* rl_up  = uc_up + (size_t)B_ * OUTW;                  // [B,L,192]
  float* erc_up = rl_up + (size_t)B_ * L_ * OUTW;             // [B,K]

  float* ws = (float*)d_ws;
  float* colsum  = ws; ws += (size_t)B_ * K_;
  float* rowsum  = ws; ws += (size_t)B_ * L_;
  float* max1    = ws; ws += (size_t)B_ * D_;
  float* max2    = ws; ws += (size_t)B_ * D_;
  float* mean_uk = ws; ws += (size_t)B_ * K_ * D_;
  float* mean_rl = ws; ws += (size_t)B_ * K_ * D_;
  float* mean_uc = ws; ws += (size_t)B_ * D_;
  float* scores  = ws; ws += (size_t)B_ * K_;

  rowsum_kernel<<<B_ * L_ / 8, 256, 0, stream>>>(e, rowsum);
  colsum_kernel<<<B_ * K_ / 256, 256, 0, stream>>>(e, colsum);
  top2_kernel<<<B_ * D_ / 256, 256, 0, stream>>>(uk, max1, max2);

  mean_gemm<true><<<dim3(K_ / 64, B_), 256, 0, stream>>>(e, rl, colsum, mean_uk);
  mean_gemm<false><<<dim3(L_ / 64, B_), 256, 0, stream>>>(e, uk, rowsum, mean_rl);
  mean_uc_kernel<<<B_ * D_ / 256, 256, 0, stream>>>(uk, erc, mean_uc);

  MlpP pfu = {uk, mean_uk, max1, max2, fu_w1, fu_b1, fu_w2, fu_b2, uk_up, OUTW, CH_};
  mlp2<0, 192, 256, 128><<<B_ * K_ / 64, 256, 0, stream>>>(pfu);
  MlpP pf  = {rl, mean_rl, nullptr, nullptr, f_w1, f_b1, f_w2, f_b2, rl_up, OUTW, CH_};
  mlp2<1, 128, 256, 128><<<B_ * L_ / 64, 256, 0, stream>>>(pf);
  MlpP pfc = {mean_uc, uc, nullptr, nullptr, fc_w1, fc_b1, fc_w2, fc_b2, uc_up, OUTW, CH_};
  mlp2<2, 128, 256, 128><<<B_ / 64, 256, 0, stream>>>(pfc);

  concat_copy_kernel<<<(B_ * K_ * D_ / 2 + 255) / 256, 256, 0, stream>>>(uk, uk_up, OUTW, CH_, B_ * K_ * D_ / 2);
  concat_copy_kernel<<<(B_ * L_ * D_ / 2 + 255) / 256, 256, 0, stream>>>(rl, rl_up, OUTW, CH_, B_ * L_ * D_ / 2);
  concat_copy_kernel<<<(B_ * D_ / 2 + 255) / 256, 256, 0, stream>>>(uc, uc_up, OUTW, CH_, B_ * D_ / 2);

  MlpP peu = {uc_up, uk_up, nullptr, nullptr, eu_w1, eu_b1, eu_w2, eu_b2, scores, 1, 1};
  mlp2<3, 192, 128, 16><<<B_ * K_ / 64, 256, 0, stream>>>(peu);
  softmax_kernel<<<B_, 256, 0, stream>>>(scores, erc_up);
}
